// CustomMHA_87711822119046
// MI455X (gfx1250) — matrix operand order repaired and checked by reading, not compiled, not run on hardware
//
#include <hip/hip_runtime.h>
#include <hip/hip_bf16.h>
#include <stdint.h>

// Problem constants (compile-time, from the reference)
#define BSZ   4
#define SEQ   2048
#define EMBD  1024
#define NHEAD 16
#define HD    64                 // head dim
#define BH    (BSZ * NHEAD)      // 64 "heads" after the buggy reshape
#define TOK   (BSZ * SEQ)        // 8192 rows in projection GEMMs
#define NELEM (TOK * EMBD)       // 8,388,608 elements per activation tensor
#define TSTRIDE (BH * HD)        // 4096: flat stride of t in the head view

typedef __attribute__((ext_vector_type(16))) __bf16 v16bf;
typedef __attribute__((ext_vector_type(8)))  float  v8f;

union Frag { v16bf v; uint4 u[2]; };   // 32 bytes: one WMMA bf16 A/B operand

__device__ __forceinline__ v8f wmma_bf16(const Frag& a, const Frag& b, v8f c) {
    // D = A(16x32) * B(32x16) + C, fp32 accumulate
    return __builtin_amdgcn_wmma_f32_16x16x32_bf16(
        false, a.v, false, b.v, (short)0, c, false, false);
}

// ---------------------------------------------------------------------------
// fp32 -> bf16 elementwise convert
// ---------------------------------------------------------------------------
__global__ void cvt_bf16_kernel(const float* __restrict__ in,
                                __bf16* __restrict__ out, int n) {
    int i = blockIdx.x * blockDim.x + threadIdx.x;
    if (i < n) out[i] = (__bf16)in[i];
}

// ---------------------------------------------------------------------------
// C[m][n] = sum_k A[m][k] * W[n][k] + bias[n]
//   A: M x K bf16 row-major (K innermost)
//   W: N x K bf16 row-major (torch Linear layout -> WMMA B columns contiguous)
// MODE 0: store bf16 at C[m*N+n]               (K projection)
// MODE 1: store bf16 scattered to vT[bh][d][s] (V projection, per-head layout)
// MODE 2: store fp32 at C[m*N+n]               (output projection)
// Block: 256 threads = 8 waves; tile 128(M) x 64(N); wave: 16 rows x 64 cols.
// 4 independent accumulators per k-step -> no WMMA hazard NOPs.
// ---------------------------------------------------------------------------
template <int MODE>
__global__ void gemm_kernel(const __bf16* __restrict__ A,
                            const __bf16* __restrict__ W,
                            const float*  __restrict__ bias,
                            __bf16* __restrict__ outb,
                            float*  __restrict__ outf,
                            int M, int N, int K) {
    const int lane  = threadIdx.x & 31;
    const int wave  = threadIdx.x >> 5;      // 0..7
    const int lhalf = lane >> 4;             // 0|1
    const int lm    = lane & 15;
    const int m0    = blockIdx.x * 128 + wave * 16;
    const int n0    = blockIdx.y * 64;

    v8f acc[4] = {};
    const __bf16* aptr = A + (size_t)(m0 + lm) * K;

    for (int k0 = 0; k0 < K; k0 += 32) {
        Frag a;
        const int ak = k0 + lhalf * 8;       // A: lane<16 K{0..7,16..23}, else +8
        a.u[0] = *(const uint4*)(aptr + ak);
        a.u[1] = *(const uint4*)(aptr + ak + 16);
        if (k0 + 32 < K)                     // stream-ahead on the HBM operand
            __builtin_prefetch(aptr + k0 + 32 + lhalf * 8, 0, 0);
#pragma unroll
        for (int nt = 0; nt < 4; ++nt) {
            Frag b;                          // B col = n0+nt*16+lm, K = k0+lhalf*8.. and +16
            const __bf16* bptr = W + (size_t)(n0 + nt * 16 + lm) * K + k0 + lhalf * 8;
            b.u[0] = *(const uint4*)(bptr);
            b.u[1] = *(const uint4*)(bptr + 16);
            acc[nt] = wmma_bf16(a, b, acc[nt]);
        }
    }

#pragma unroll
    for (int nt = 0; nt < 4; ++nt) {
#pragma unroll
        for (int r = 0; r < 8; ++r) {
            const int mm = m0 + r + lhalf * 8;
            const int nn = n0 + nt * 16 + lm;
            const float v = acc[nt][r] + bias[nn];
            if (MODE == 2) {
                outf[(size_t)mm * N + nn] = v;
            } else if (MODE == 0) {
                outb[(size_t)mm * N + nn] = (__bf16)v;
            } else {
                // flat index f = mm*1024+nn viewed as [s][bh][d] (buggy reshape)
                const int sp = mm >> 2;
                const int bh = ((mm & 3) << 4) + (nn >> 6);
                const int d  = nn & 63;
                outb[(size_t)bh * (HD * SEQ) + (size_t)d * SEQ + sp] = (__bf16)v;
            }
        }
    }
}

// ---------------------------------------------------------------------------
// Attention: one workgroup per (16 q-rows, head). 8 waves.
//   phase 1: logits[16][2048] = q_tile @ k^T -> LDS fp32 (paired s-tiles,
//            two independent WMMA chains to avoid D->C hazard NOPs)
//   phase 2: row softmax (wave reductions), coalesced fp32 attn store,
//            bf16 probabilities kept in LDS
//   phase 3: ctx = P @ V, K split across wave pairs, two interleaved
//            accumulators, LDS reduce
// Dynamic LDS: 16*2048*4 (logits/reduce) + 16*2048*2 (bf16 P) = 192 KB
// ---------------------------------------------------------------------------
__global__ void attn_kernel(const __bf16* __restrict__ q,    // flat t*4096+bh*64+d
                            const __bf16* __restrict__ kp,   // flat (projected K)
                            const __bf16* __restrict__ vT,   // [bh][d][s]
                            float* __restrict__ attn_out,    // [bh][t][s] fp32
                            __bf16* __restrict__ ctx) {      // flat bf16
    extern __shared__ char smem[];
    float*  logits = (float*)smem;                       // [16][SEQ]
    __bf16* pmat   = (__bf16*)(smem + 16 * SEQ * 4);     // [16][SEQ]

    const int tb    = blockIdx.x;
    const int bh    = blockIdx.y;
    const int t0    = tb * 16;
    const int lane  = threadIdx.x & 31;
    const int wave  = threadIdx.x >> 5;
    const int lhalf = lane >> 4;
    const int lm    = lane & 15;

    // ---- Q fragments (head_dim=64 -> two K-steps); each wave keeps a copy ----
    Frag a0, a1;
    {
        const __bf16* qrow = q + (size_t)(t0 + lm) * TSTRIDE + bh * HD;
        const int kb = lhalf * 8;
        a0.u[0] = *(const uint4*)(qrow + kb);
        a0.u[1] = *(const uint4*)(qrow + kb + 16);
        a1.u[0] = *(const uint4*)(qrow + 32 + kb);
        a1.u[1] = *(const uint4*)(qrow + 32 + kb + 16);
    }

    // ---- phase 1: each wave covers 16 of the 128 s-tiles, two at a time ----
    for (int st = wave * 16; st < wave * 16 + 16; st += 2) {
        const int s0 = st * 16;
        Frag b00, b01, b10, b11;  // B[d][s]: column s = kp row (d innermost)
        const __bf16* krow0 = kp + (size_t)(s0 + lm) * TSTRIDE + bh * HD + lhalf * 8;
        const __bf16* krow1 = krow0 + (size_t)16 * TSTRIDE;
        b00.u[0] = *(const uint4*)(krow0);
        b00.u[1] = *(const uint4*)(krow0 + 16);
        b01.u[0] = *(const uint4*)(krow0 + 32);
        b01.u[1] = *(const uint4*)(krow0 + 48);
        b10.u[0] = *(const uint4*)(krow1);
        b10.u[1] = *(const uint4*)(krow1 + 16);
        b11.u[0] = *(const uint4*)(krow1 + 32);
        b11.u[1] = *(const uint4*)(krow1 + 48);
        v8f c0 = {}, c1 = {};
        c0 = wmma_bf16(a0, b00, c0);     // independent chains interleave ->
        c1 = wmma_bf16(a0, b10, c1);     // hazard slots filled with real work
        c0 = wmma_bf16(a1, b01, c0);
        c1 = wmma_bf16(a1, b11, c1);
#pragma unroll
        for (int r = 0; r < 8; ++r) {
            logits[(r + lhalf * 8) * SEQ + (s0 + lm)]      = c0[r];
            logits[(r + lhalf * 8) * SEQ + (s0 + 16 + lm)] = c1[r];
        }
    }
    __syncthreads();

    // ---- phase 2: softmax, wave w owns rows 2w, 2w+1 ----
    float rmax[2], rinv[2];
#pragma unroll
    for (int rr = 0; rr < 2; ++rr) {
        const int row = wave * 2 + rr;
        float m = -3.0e38f;
        for (int j = lane; j < SEQ; j += 32)
            m = fmaxf(m, logits[row * SEQ + j]);
        for (int off = 16; off > 0; off >>= 1)
            m = fmaxf(m, __shfl_xor(m, off, 32));
        float s = 0.f;
        for (int j = lane; j < SEQ; j += 32)
            s += __expf(logits[row * SEQ + j] - m);
        for (int off = 16; off > 0; off >>= 1)
            s += __shfl_xor(s, off, 32);
        rmax[rr] = m;
        rinv[rr] = 1.f / s;
    }
#pragma unroll
    for (int rr = 0; rr < 2; ++rr) {
        const int row = wave * 2 + rr;
        const size_t gbase = (size_t)bh * SEQ * SEQ + (size_t)(t0 + row) * SEQ;
        for (int j = lane; j < SEQ; j += 32) {          // coalesced
            const float p = __expf(logits[row * SEQ + j] - rmax[rr]) * rinv[rr];
            attn_out[gbase + j] = p;
            pmat[row * SEQ + j] = (__bf16)p;
        }
    }
    __syncthreads();

    // ---- phase 3: ctx(16x64) = P(16x2048) @ V(2048x64) ----
    const int nt = wave & 3;                 // d-tile (16 cols)
    const int kh = wave >> 2;                // K half
    const int d0 = nt * 16;
    v8f c0 = {}, c1 = {};
    const int sbeg = kh * (SEQ / 2);
    const __bf16* vbase = vT + (size_t)bh * (HD * SEQ) +
                          (size_t)(d0 + lm) * SEQ + lhalf * 8;
    const __bf16* pbase = pmat + (size_t)lm * SEQ + lhalf * 8;
    for (int s0 = sbeg; s0 < sbeg + SEQ / 2; s0 += 64) {
        Frag aA, bA, aB, bB;
        aA.u[0] = *(const uint4*)(pbase + s0);
        aA.u[1] = *(const uint4*)(pbase + s0 + 16);
        bA.u[0] = *(const uint4*)(vbase + s0);
        bA.u[1] = *(const uint4*)(vbase + s0 + 16);
        aB.u[0] = *(const uint4*)(pbase + s0 + 32);
        aB.u[1] = *(const uint4*)(pbase + s0 + 48);
        bB.u[0] = *(const uint4*)(vbase + s0 + 32);
        bB.u[1] = *(const uint4*)(vbase + s0 + 48);
        c0 = wmma_bf16(aA, bA, c0);          // two independent chains
        c1 = wmma_bf16(aB, bB, c1);
    }
    v8f c = c0 + c1;

    float* red = (float*)smem;               // logits region is free now
    if (kh == 1) {
#pragma unroll
        for (int r = 0; r < 8; ++r)
            red[nt * 256 + r * 32 + lane] = c[r];
    }
    __syncthreads();
    if (kh == 0) {
#pragma unroll
        for (int r = 0; r < 8; ++r) {
            const float v = c[r] + red[nt * 256 + r * 32 + lane];
            const int trow = t0 + r + lhalf * 8;
            ctx[(size_t)trow * TSTRIDE + bh * HD + d0 + lm] = (__bf16)v;
        }
    }
}

// ---------------------------------------------------------------------------
extern "C" void kernel_launch(void* const* d_in, const int* in_sizes, int n_in,
                              void* d_out, int out_size, void* d_ws, size_t ws_size,
                              hipStream_t stream) {
    const float* query   = (const float*)d_in[0];
    const float* key     = (const float*)d_in[1];
    const float* value   = (const float*)d_in[2];
    const float* key_w   = (const float*)d_in[3];
    const float* key_b   = (const float*)d_in[4];
    const float* value_w = (const float*)d_in[5];
    const float* value_b = (const float*)d_in[6];
    const float* out_w   = (const float*)d_in[7];
    const float* out_b   = (const float*)d_in[8];
    (void)in_sizes; (void)n_in; (void)out_size; (void)ws_size;

    char* ws = (char*)d_ws;
    const size_t MB = 1024u * 1024u;
    __bf16* qb   = (__bf16*)(ws + 0 * MB);    // 16 MB: query bf16
    __bf16* kinb = (__bf16*)(ws + 16 * MB);   // 16 MB: key input bf16
    __bf16* vinb = (__bf16*)(ws + 32 * MB);   // 16 MB: value input bf16
    __bf16* kwb  = (__bf16*)(ws + 48 * MB);   //  2 MB: key_w bf16
    __bf16* vwb  = (__bf16*)(ws + 50 * MB);   //  2 MB: value_w bf16
    __bf16* owb  = (__bf16*)(ws + 52 * MB);   //  2 MB: out_w bf16
    __bf16* kpb  = (__bf16*)(ws + 54 * MB);   // 16 MB: projected K (flat)
    __bf16* vTb  = (__bf16*)(ws + 70 * MB);   // 16 MB: projected V [bh][d][s]
    __bf16* ctxb = (__bf16*)(ws + 86 * MB);   // 16 MB: attn@v result (flat)

    float* out_proj = (float*)d_out;                       // (4,2048,1024) fp32
    float* attn_out = (float*)d_out + (size_t)NELEM;       // (64,2048,2048) fp32

    cvt_bf16_kernel<<<NELEM / 256, 256, 0, stream>>>(query, qb, NELEM);
    cvt_bf16_kernel<<<NELEM / 256, 256, 0, stream>>>(key, kinb, NELEM);
    cvt_bf16_kernel<<<NELEM / 256, 256, 0, stream>>>(value, vinb, NELEM);
    cvt_bf16_kernel<<<(EMBD * EMBD) / 256, 256, 0, stream>>>(key_w, kwb, EMBD * EMBD);
    cvt_bf16_kernel<<<(EMBD * EMBD) / 256, 256, 0, stream>>>(value_w, vwb, EMBD * EMBD);
    cvt_bf16_kernel<<<(EMBD * EMBD) / 256, 256, 0, stream>>>(out_w, owb, EMBD * EMBD);

    dim3 gproj(TOK / 128, EMBD / 64);
    gemm_kernel<0><<<gproj, 256, 0, stream>>>(kinb, kwb, key_b,   kpb, nullptr, TOK, EMBD, EMBD);
    gemm_kernel<1><<<gproj, 256, 0, stream>>>(vinb, vwb, value_b, vTb, nullptr, TOK, EMBD, EMBD);

    dim3 gattn(SEQ / 16, BH);
    const size_t lds_bytes = 16 * SEQ * 4 + 16 * SEQ * 2;  // 192 KB (< 320 KB/WG)
    attn_kernel<<<gattn, 256, lds_bytes, stream>>>(qb, kpb, vTb, attn_out, ctxb);

    gemm_kernel<2><<<gproj, 256, 0, stream>>>(ctxb, owb, out_b, nullptr, out_proj, TOK, EMBD, EMBD);
}
